// ByteToLatentAttention_30502857736797
// MI455X (gfx1250) — matrix operand order repaired and checked by reading, not compiled, not run on hardware
//
#include <hip/hip_runtime.h>
#include <hip/hip_bf16.h>
#include <cmath>

typedef _Float16 half_t;
typedef __attribute__((ext_vector_type(16))) _Float16 v16h;
typedef __attribute__((ext_vector_type(8)))  _Float16 v8h;
typedef __attribute__((ext_vector_type(8)))  float    v8f;

constexpr int BB   = 2;
constexpr int SS   = 4096;
constexpr int DD   = 512;
constexpr int BPL  = 4;
constexpr int HH   = 16;
constexpr int DQK  = 64;
constexpr int DLAT = 1024;
constexpr int LQ   = SS / BPL;    // 1024
constexpr int HD   = HH * DQK;    // 1024
constexpr int MK   = BPL * DD;    // 2048
constexpr float EPSF = 1.1920929e-07f;
constexpr float ROPE_BASE = 10000.0f;

// Packed fragment layouts (wave32, 16-bit WMMA, CDNA5 ISA):
//  A frag (16x32): lane = m + 16*hiA, hiA=(kl>>3)&1; e in 0..7 <-> kl = e + hiA*8,
//                  e in 8..15 <-> kl = 16+(e-8)+hiA*8
//  B frag (32x16): lane = n + 16*hi, elem e <-> kl as in A frag with hiA = hi
// Each lane's 16 halves contiguous (32B) -> frag read = 2x ds_load_b128.

// CDNA5 async memory->LDS copy, 16B per lane, tracked by ASYNCcnt
// (CDNA5 ISA).  LDS byte address = low 32 bits of the
// generic pointer (ISA §10.2 aperture).
__device__ inline void async_b128(void* lds, const void* gmem) {
  asm volatile("global_load_async_to_lds_b128 %0, %1, off"
               :: "v"((unsigned)(unsigned long long)lds), "v"(gmem)
               : "memory");
}

// ---------------------------------------------------------------------------
// RMSNorm: one block per row (B*S rows, D=512) -> f16 normalized + f16 raw x
// ---------------------------------------------------------------------------
__global__ __launch_bounds__(256) void rmsnorm_kernel(
    const float* __restrict__ x, const float* __restrict__ w,
    half_t* __restrict__ xn, half_t* __restrict__ xr)
{
  int row = blockIdx.x;
  size_t base = (size_t)row * DD;
  int t = threadIdx.x;
  float v0 = x[base + t];
  float v1 = x[base + 256 + t];
  float ss = v0 * v0 + v1 * v1;
#pragma unroll
  for (int off = 16; off >= 1; off >>= 1) ss += __shfl_xor(ss, off);
  __shared__ float red[8];
  if ((t & 31) == 0) red[t >> 5] = ss;
  __syncthreads();
  float tot = 0.f;
#pragma unroll
  for (int i = 0; i < 8; ++i) tot += red[i];
  float rs = rsqrtf(tot * (1.0f / (float)DD) + EPSF);
  xn[base + t]       = (half_t)(v0 * rs * w[t]);
  xn[base + 256 + t] = (half_t)(v1 * rs * w[256 + t]);
  xr[base + t]       = (half_t)v0;
  xr[base + 256 + t] = (half_t)v1;
}

// f32 (KxN row-major) -> f16 transposed (NxK): GEMM B operands become N-major
__global__ void f32_to_f16_t_kernel(const float* __restrict__ in,
                                    half_t* __restrict__ out, int K, int N)
{
  int i = blockIdx.x * blockDim.x + threadIdx.x;
  if (i >= K * N) return;
  int k = i / N, n = i - k * N;
  out[(size_t)n * K + k] = (half_t)in[i];
}

// ---------------------------------------------------------------------------
// Tiled f16 WMMA GEMM: C(MxN,f32) = A(MxK,f16,row-major) @ Bt(NxK,f16)^T
// Block tile 64(M) x 128(N), 8 waves, K-step 64.
// Double-buffered LDS fed by async-to-LDS DMA; frag reads are ds_load_b128.
// ---------------------------------------------------------------------------
constexpr int GT_BM = 64;
constexpr int GT_BN = 128;
constexpr int GT_BK = 64;

__global__ __launch_bounds__(256) void gemm_f16_wmma_kernel(
    const half_t* __restrict__ A, const half_t* __restrict__ Bt,
    const float* __restrict__ bias, float* __restrict__ C,
    int M, int N, int K, int accumulate)
{
  __shared__ __attribute__((aligned(32))) half_t As[2][4][2][512];  // [buf][wm][chunk]
  __shared__ __attribute__((aligned(32))) half_t Bs[2][8][2][512];  // [buf][nf][chunk]
  int tid  = threadIdx.x;
  int wave = tid >> 5;
  int lane = tid & 31;
  int wm = wave & 3;
  int wn = wave >> 2;
  int block_m = blockIdx.y * GT_BM;
  int block_n = blockIdx.x * GT_BN;

  // per-thread staging coordinates
  int ar = tid >> 2;             // A row 0..63
  int ac = (tid & 3) * 16;       // A k-offset 0/16/32/48
  int nl = tid >> 1;             // B n-row 0..127
  int cb = (tid & 1) * 32;       // B k-offset 0/32

  auto stage = [&](int k0, int bi) {
    // A tile 64x64 -> packed frags, two 16B async granules
    const half_t* gA = A + (size_t)(block_m + ar) * K + k0 + ac;
    int chA = ac >> 5;
#pragma unroll
    for (int seg = 0; seg < 2; ++seg) {
      int kl = (ac & 31) + seg * 8;
      int hiA = (kl >> 3) & 1;
      int e0  = (kl & 16) ? 8 : 0;
      async_b128(&As[bi][ar >> 4][chA][((ar & 15) + 16 * hiA) * 16 + e0], gA + seg * 8);
    }
    // B tile (Bt N-major) 128n x 64k -> packed frags, four 16B async granules
    const half_t* gB = Bt + (size_t)(block_n + nl) * K + k0 + cb;
    int chB = cb >> 5;
    unsigned l0 = (nl & 15) * 16;          // hi=0 half (k = cb..cb+7, cb+16..cb+23)
    unsigned l1 = ((nl & 15) + 16) * 16;   // hi=1 half (k = cb+8..cb+15, cb+24..cb+31)
    async_b128(&Bs[bi][nl >> 4][chB][l0],     gB);
    async_b128(&Bs[bi][nl >> 4][chB][l0 + 8], gB + 16);
    async_b128(&Bs[bi][nl >> 4][chB][l1],     gB + 8);
    async_b128(&Bs[bi][nl >> 4][chB][l1 + 8], gB + 24);
  };

  v8f acc[4] = {};
  int nt = K / GT_BK;
  stage(0, 0);

  for (int t = 0; t < nt; ++t) {
    int bi = t & 1;
    if (t + 1 < nt) {
      stage((t + 1) * GT_BK, (t + 1) & 1);          // overlap DMA with WMMA
      asm volatile("s_wait_asynccnt 0x6" ::: "memory");  // tile t done (in-order)
    } else {
      asm volatile("s_wait_asynccnt 0x0" ::: "memory");
    }
    __syncthreads();

#pragma unroll
    for (int ch = 0; ch < 2; ++ch) {
      v16h af = *(const v16h*)&As[bi][wm][ch][lane * 16];
#pragma unroll
      for (int f = 0; f < 4; ++f) {
        v16h bf = *(const v16h*)&Bs[bi][wn * 4 + f][ch][lane * 16];
        acc[f] = __builtin_amdgcn_wmma_f32_16x16x32_f16(
            false, af, false, bf, (short)0, acc[f], false, false);
      }
    }
    __syncthreads();   // all reads of buf bi done before it is re-staged
  }

  int hi = lane >> 4;
  int n_in = lane & 15;
#pragma unroll
  for (int f = 0; f < 4; ++f) {
    int gn = block_n + wn * 64 + f * 16 + n_in;
#pragma unroll
    for (int r = 0; r < 8; ++r) {
      int gm = block_m + wm * 16 + hi * 8 + r;
      float v = acc[f][r];
      if (bias) v += bias[gn];
      size_t idx = (size_t)gm * N + gn;
      if (accumulate) v += C[idx];
      C[idx] = v;
    }
  }
}

// ---------------------------------------------------------------------------
// RoPE + head split.
// q32 (B*LQ, H*64) -> qhP: A-fragment-packed global layout
//   [b][h][q16tile][chunk(2)][lane(32)*16+e], pos = q*BPL
// ---------------------------------------------------------------------------
__device__ inline size_t q_packed_idx(int b, int h, int q, int d) {
  int qt16 = q >> 4, m = q & 15;
  int chunk = d >> 5, kl = d & 31;
  int hiA = (kl >> 3) & 1;
  int e = (kl & 16) ? (8 + (kl & 7)) : (kl & 7);
  return ((((size_t)b * HH + h) * (LQ / 16) + qt16) * 2 + chunk) * 512
         + (m + 16 * hiA) * 16 + e;
}

__global__ void rope_q_kernel(const float* __restrict__ q32, half_t* __restrict__ qhP)
{
  int i = blockIdx.x * blockDim.x + threadIdx.x;  // B*LQ*H*32 = 2^20
  if (i >= BB * LQ * HH * 32) return;
  int j = i & 31;
  int h = (i >> 5) & 15;
  int q = (i >> 9) & (LQ - 1);
  int b = i >> 19;
  float inv = powf(ROPE_BASE, -(float)j / 32.0f);
  float ang = (float)(q * BPL) * inv;
  float c = cosf(ang), s = sinf(ang);
  size_t src = ((size_t)b * LQ + q) * HD + h * DQK + j;
  float x1 = q32[src], x2 = q32[src + 32];
  qhP[q_packed_idx(b, h, q, j)]      = (half_t)(x1 * c - x2 * s);
  qhP[q_packed_idx(b, h, q, j + 32)] = (half_t)(x1 * s + x2 * c);
}

// kv32 (B*S, 2*H*64) -> kh (roped, row-major [b][h][s][d]) and
// vT (d-major [b][h][d][s]) so attention staging is 32B vector granules.
__global__ void rope_kv_kernel(const float* __restrict__ kv32,
                               half_t* __restrict__ kh, half_t* __restrict__ vT)
{
  int i = blockIdx.x * blockDim.x + threadIdx.x;  // B*S*H*32 = 2^22
  if (i >= BB * SS * HH * 32) return;
  int j = i & 31;
  int h = (i >> 5) & 15;
  int s = (i >> 9) & (SS - 1);
  int b = i >> 21;
  float inv = powf(ROPE_BASE, -(float)j / 32.0f);
  float ang = (float)s * inv;
  float cs = cosf(ang), sn = sinf(ang);
  size_t src = ((size_t)b * SS + s) * (2 * HD) + h * DQK + j;
  float x1 = kv32[src], x2 = kv32[src + 32];
  size_t kdst = (((size_t)b * HH + h) * SS + s) * DQK + j;
  kh[kdst]      = (half_t)(x1 * cs - x2 * sn);
  kh[kdst + 32] = (half_t)(x1 * sn + x2 * cs);
  float v1 = kv32[src + HD], v2 = kv32[src + HD + 32];
  size_t vbase = ((size_t)b * HH + h) * DQK * SS;
  vT[vbase + (size_t)j * SS + s]        = (half_t)v1;
  vT[vbase + (size_t)(j + 32) * SS + s] = (half_t)v2;
}

// ---------------------------------------------------------------------------
// Flash attention: 8 waves, 128 q rows/block, 64-key tiles,
// double-buffered async K/V staging, packed LDS frags.
// ---------------------------------------------------------------------------
constexpr int AT_QT = 128;
constexpr int AT_KT = 64;

__global__ __launch_bounds__(256) void flash_attn_kernel(
    const half_t* __restrict__ qhP, const half_t* __restrict__ kh,
    const half_t* __restrict__ vT, half_t* __restrict__ ah)
{
  __shared__ __attribute__((aligned(32))) half_t Ks[2][4][2][512];  // [buf][keyf][dchunk]
  __shared__ __attribute__((aligned(32))) half_t Vs[2][4][2][512];  // [buf][dgrp][keychunk]
  __shared__ __attribute__((aligned(32))) half_t Ps[8][2][512];     // [wave][keychunk]

  int tid  = threadIdx.x;
  int wave = tid >> 5;
  int lane = tid & 31;
  int hi   = lane >> 4;
  int n_in = lane & 15;

  constexpr int QTILES = LQ / AT_QT;  // 8
  int qt = blockIdx.x % QTILES;
  int bh = blockIdx.x / QTILES;
  int h  = bh % HH;
  int b  = bh / HH;

  const half_t* qp = qhP
      + ((((size_t)b * HH + h) * (LQ / 16) + (qt * (AT_QT / 16) + wave)) * 2) * 512;
  v16h aq0 = *(const v16h*)(qp + lane * 16);
  v16h aq1 = *(const v16h*)(qp + 512 + lane * 16);

  float mrow[8], lrow[8];
  v8f oacc[4] = {};
#pragma unroll
  for (int r = 0; r < 8; ++r) { mrow[r] = -3.0e38f; lrow[r] = 0.f; }

  const half_t* kbase  = kh + ((size_t)b * HH + h) * SS * DQK;
  const half_t* vtbase = vT + ((size_t)b * HH + h) * DQK * SS;

  int sr = tid >> 2;             // key row (for K) / d row (for V), 0..63
  int sc = (tid & 3) * 16;       // 16-half column offset

  auto stageKV = [&](int s0, int bi) {
    const half_t* gk = kbase + (size_t)(s0 + sr) * DQK + sc;
    unsigned lk = (sr & 15) * 16 + 8 * ((sc >> 4) & 1);
    async_b128(&Ks[bi][sr >> 4][sc >> 5][lk],       gk);
    async_b128(&Ks[bi][sr >> 4][sc >> 5][lk + 256], gk + 8);
    const half_t* gv = vtbase + (size_t)sr * SS + s0 + sc;
    async_b128(&Vs[bi][sr >> 4][sc >> 5][lk],       gv);
    async_b128(&Vs[bi][sr >> 4][sc >> 5][lk + 256], gv + 8);
  };

  int nt = SS / AT_KT;  // 64
  stageKV(0, 0);

  for (int t = 0; t < nt; ++t) {
    int bi = t & 1;
    if (t + 1 < nt) {
      stageKV((t + 1) * AT_KT, (t + 1) & 1);
      asm volatile("s_wait_asynccnt 0x4" ::: "memory");
    } else {
      asm volatile("s_wait_asynccnt 0x0" ::: "memory");
    }
    __syncthreads();

    // scores: four 16x16 C frags over 64 keys, K-dim = d (2 chunks)
    v8f sfr[4];
#pragma unroll
    for (int f = 0; f < 4; ++f) {
      v8f c = {};
      v16h b0 = *(const v16h*)&Ks[bi][f][0][lane * 16];
      c = __builtin_amdgcn_wmma_f32_16x16x32_f16(false, aq0, false, b0,
                                                 (short)0, c, false, false);
      v16h b1 = *(const v16h*)&Ks[bi][f][1][lane * 16];
      c = __builtin_amdgcn_wmma_f32_16x16x32_f16(false, aq1, false, b1,
                                                 (short)0, c, false, false);
#pragma unroll
      for (int e = 0; e < 8; ++e) c[e] *= 0.125f;  // 1/sqrt(64)
      sfr[f] = c;
    }

    // online softmax (rows in 16-lane halves; xor-reduce 8,4,2,1)
#pragma unroll
    for (int r = 0; r < 8; ++r) {
      float mx = -3.0e38f;
#pragma unroll
      for (int f = 0; f < 4; ++f) mx = fmaxf(mx, sfr[f][r]);
#pragma unroll
      for (int o = 8; o >= 1; o >>= 1) mx = fmaxf(mx, __shfl_xor(mx, o));
      float nm = fmaxf(mrow[r], mx);
      float alpha = __expf(mrow[r] - nm);
      mrow[r] = nm;
      float rsum = 0.f;
#pragma unroll
      for (int f = 0; f < 4; ++f) {
        float p = __expf(sfr[f][r] - nm);
        sfr[f][r] = p;
        rsum += p;
      }
#pragma unroll
      for (int o = 8; o >= 1; o >>= 1) rsum += __shfl_xor(rsum, o);
      lrow[r] = lrow[r] * alpha + rsum;
#pragma unroll
      for (int g = 0; g < 4; ++g) oacc[g][r] *= alpha;
    }

    // stage P (C-layout -> packed A-layout), per-wave region
#pragma unroll
    for (int f = 0; f < 4; ++f) {
      int chunk = f >> 1;
      int kl = (f & 1) * 16 + n_in;
      int hiA = (kl >> 3) & 1;
      int e = (kl & 16) ? (8 + (kl & 7)) : (kl & 7);
#pragma unroll
      for (int r = 0; r < 8; ++r)
        Ps[wave][chunk][((hi * 8 + r) + 16 * hiA) * 16 + e] = (half_t)sfr[f][r];
    }
    asm volatile("s_wait_dscnt 0" ::: "memory");  // wave-local DS ordering

    // O += P @ V
    v16h ap0 = *(const v16h*)&Ps[wave][0][lane * 16];
    v16h ap1 = *(const v16h*)&Ps[wave][1][lane * 16];
#pragma unroll
    for (int g = 0; g < 4; ++g) {
      v16h bv0 = *(const v16h*)&Vs[bi][g][0][lane * 16];
      oacc[g] = __builtin_amdgcn_wmma_f32_16x16x32_f16(false, ap0, false, bv0,
                                                       (short)0, oacc[g], false, false);
      v16h bv1 = *(const v16h*)&Vs[bi][g][1][lane * 16];
      oacc[g] = __builtin_amdgcn_wmma_f32_16x16x32_f16(false, ap1, false, bv1,
                                                       (short)0, oacc[g], false, false);
    }
    __syncthreads();
  }

  // normalize + store attn_concat (B, LQ, H*64) f16
#pragma unroll
  for (int g = 0; g < 4; ++g)
#pragma unroll
    for (int r = 0; r < 8; ++r) {
      int qrow = qt * AT_QT + wave * 16 + hi * 8 + r;
      int col = h * DQK + g * 16 + n_in;
      ah[((size_t)b * LQ + qrow) * HD + col] = (half_t)(oacc[g][r] / lrow[r]);
    }
}

// ---------------------------------------------------------------------------
extern "C" void kernel_launch(void* const* d_in, const int* in_sizes, int n_in,
                              void* d_out, int out_size, void* d_ws, size_t ws_size,
                              hipStream_t stream) {
  const float* x      = (const float*)d_in[0];
  const float* norm_w = (const float*)d_in[1];
  const float* wq_w   = (const float*)d_in[2];
  const float* wq_b   = (const float*)d_in[3];
  const float* wkv_w  = (const float*)d_in[4];
  const float* wkv_b  = (const float*)d_in[5];
  const float* wout_w = (const float*)d_in[6];
  const float* wout_b = (const float*)d_in[7];
  const float* wbyp_w = (const float*)d_in[8];
  float* out = (float*)d_out;
  (void)in_sizes; (void)n_in; (void)out_size; (void)ws_size;

  char* ws = (char*)d_ws;
  size_t off = 0;
  auto alloc = [&](size_t bytes) {
    void* p = ws + off;
    off = (off + bytes + 255) & ~(size_t)255;
    return p;
  };

  half_t* xn16   = (half_t*)alloc((size_t)BB * SS * DD * sizeof(half_t));
  half_t* xr16   = (half_t*)alloc((size_t)BB * SS * DD * sizeof(half_t));
  half_t* wq16t  = (half_t*)alloc((size_t)MK * HD * sizeof(half_t));       // [n][k]
  half_t* wkv16t = (half_t*)alloc((size_t)DD * 2 * HD * sizeof(half_t));   // [n][k]
  half_t* wout16t= (half_t*)alloc((size_t)HD * DLAT * sizeof(half_t));     // [n][k]
  half_t* wbyp16t= (half_t*)alloc((size_t)MK * DLAT * sizeof(half_t));     // [n][k]
  float*  q32    = (float*)alloc((size_t)BB * LQ * HD * sizeof(float));
  float*  kv32   = (float*)alloc((size_t)BB * SS * 2 * HD * sizeof(float));
  half_t* qhP    = (half_t*)alloc((size_t)BB * HH * LQ * DQK * sizeof(half_t));
  half_t* kh     = (half_t*)alloc((size_t)BB * HH * SS * DQK * sizeof(half_t));
  half_t* vT     = (half_t*)alloc((size_t)BB * HH * SS * DQK * sizeof(half_t));
  half_t* ah     = (half_t*)alloc((size_t)BB * LQ * HD * sizeof(half_t));

  // 1. RMSNorm + f16 conversion of x
  rmsnorm_kernel<<<BB * SS, 256, 0, stream>>>(x, norm_w, xn16, xr16);

  // 2. Weight conversions f32 -> f16 transposed (N-major)
  auto convt = [&](const float* src, half_t* dst, int K, int N) {
    int n = K * N;
    f32_to_f16_t_kernel<<<(n + 255) / 256, 256, 0, stream>>>(src, dst, K, N);
  };
  convt(wq_w,   wq16t,  MK, HD);
  convt(wkv_w,  wkv16t, DD, 2 * HD);
  convt(wout_w, wout16t, HD, DLAT);
  convt(wbyp_w, wbyp16t, MK, DLAT);

  // 3. Q projection: (B*LQ, 2048) @ (2048, 1024) + wq_b
  {
    dim3 grid(HD / GT_BN, (BB * LQ) / GT_BM);
    gemm_f16_wmma_kernel<<<grid, 256, 0, stream>>>(
        xn16, wq16t, wq_b, q32, BB * LQ, HD, MK, 0);
  }
  // 4. KV projection: (B*S, 512) @ (512, 2048) + wkv_b
  {
    dim3 grid((2 * HD) / GT_BN, (BB * SS) / GT_BM);
    gemm_f16_wmma_kernel<<<grid, 256, 0, stream>>>(
        xn16, wkv16t, wkv_b, kv32, BB * SS, 2 * HD, DD, 0);
  }

  // 5. RoPE + head split (q pre-packed to A-frag layout; v transposed)
  rope_q_kernel<<<(BB * LQ * HH * 32) / 256, 256, 0, stream>>>(q32, qhP);
  rope_kv_kernel<<<(BB * SS * HH * 32) / 256, 256, 0, stream>>>(kv32, kh, vT);

  // 6. Flash attention
  flash_attn_kernel<<<BB * HH * (LQ / AT_QT), 256, 0, stream>>>(qhP, kh, vT, ah);

  // 7. Bypass GEMM writes d_out with wout_b bias: (B*LQ, 2048) @ (2048, 1024)
  {
    dim3 grid(DLAT / GT_BN, (BB * LQ) / GT_BM);
    gemm_f16_wmma_kernel<<<grid, 256, 0, stream>>>(
        xr16, wbyp16t, wout_b, out, BB * LQ, DLAT, MK, 0);
  }
  // 8. Output GEMM accumulates: (B*LQ, 1024) @ (1024, 1024)
  {
    dim3 grid(DLAT / GT_BN, (BB * LQ) / GT_BM);
    gemm_f16_wmma_kernel<<<grid, 256, 0, stream>>>(
        ah, wout16t, nullptr, out, BB * LQ, DLAT, HD, 1);
  }
}
